// GCN_alignment_flat_47502338294131
// MI455X (gfx1250) — matrix operand order repaired and checked by reading, not compiled, not run on hardware
//
#include <hip/hip_runtime.h>
#include <hip/hip_bf16.h>
#include <math.h>

// ---------------- constants ----------------
#define V_ 8192
#define H_ 256
#define B_ 64
#define L_ 128
#define BL_ (B_ * L_)   // 8192

typedef __attribute__((ext_vector_type(16))) __bf16 v16bf;
typedef __attribute__((ext_vector_type(8)))  __bf16 v8bf;
typedef __attribute__((ext_vector_type(8)))  float  v8f;

#define WMMA_BF16(a, b, c) \
  __builtin_amdgcn_wmma_f32_16x16x32_bf16(false, (a), false, (b), (short)0, (c), false, false)

// ---------------- WMMA fragment helpers (CDNA5 16x16x32 bf16) ----------------
// All fragment loads are K-contiguous by construction -> 16B vector loads (b128).
//
// A-matrix 16x32 (MxK), 16 bf16 per lane:
//   lane l (l<16):  M=l,    elems 0..7 = K 0..7,   elems 8..15 = K 16..23
//   lane l (l>=16): M=l-16, elems 0..7 = K 8..15,  elems 8..15 = K 24..31
__device__ __forceinline__ v16bf frag_a_bf(const __bf16* __restrict__ p, int mS) {
  const int l = threadIdx.x & 31;
  const __bf16* q = p + (size_t)(l & 15) * mS + ((l >> 4) << 3);
  v8bf lo = *(const v8bf*)q;          // K kb..kb+7
  v8bf hi = *(const v8bf*)(q + 16);   // K kb+16..kb+23
  v16bf a;
#pragma unroll
  for (int i = 0; i < 8; ++i) { a[i] = lo[i]; a[8 + i] = hi[i]; }
  return a;
}

// A fragment from fp32 source, converted in-register (used to stream A once).
__device__ __forceinline__ v16bf frag_a_f32(const float* __restrict__ p, int mS) {
  const int l = threadIdx.x & 31;
  const float* q = p + (size_t)(l & 15) * mS + ((l >> 4) << 3);
  v16bf a;
#pragma unroll
  for (int i = 0; i < 8; ++i) a[i] = (__bf16)q[i];
  q += 16;
#pragma unroll
  for (int i = 0; i < 8; ++i) a[8 + i] = (__bf16)q[i];
  return a;
}

// B-matrix 32x16 (KxN), 16 bf16 per lane; source layout is [N][K] (K contiguous):
//   lane l (l<16):  N=l,    elems 0..7 = K 0..7,   elems 8..15 = K 16..23
//   lane l (l>=16): N=l-16, elems 0..7 = K 8..15,  elems 8..15 = K 24..31
__device__ __forceinline__ v16bf frag_b_bf(const __bf16* __restrict__ p, int nS) {
  const int l = threadIdx.x & 31;
  const __bf16* q = p + (size_t)(l & 15) * nS + ((l >> 4) << 3);
  v8bf lo = *(const v8bf*)q;
  v8bf hi = *(const v8bf*)(q + 16);
  v16bf b;
#pragma unroll
  for (int i = 0; i < 8; ++i) { b[i] = lo[i]; b[8 + i] = hi[i]; }
  return b;
}

// C/D 16x16 f32: VGPR r -> M = r + 8*(lane>=16), N = lane&15
// MODE: 0 = plain store; 1 = bias+relu; 2 = bias+sigmoid;
//       3 = bias+sigmoid then highway combine: out = g*h + (1-g)*x
template <int MODE>
__device__ __forceinline__ void store_c(float* __restrict__ p, int mS,
                                        const float* __restrict__ bias,
                                        const float* __restrict__ hp,
                                        const float* __restrict__ xp,
                                        v8f c) {
  const int l  = threadIdx.x & 31;
  const int n  = l & 15;
  const int mb = (l >> 4) << 3;
  float bv = 0.0f;
  if (MODE != 0) bv = bias[n];
#pragma unroll
  for (int r = 0; r < 8; ++r) {
    float v = c[r] + bv;
    size_t o = (size_t)(mb + r) * mS + n;
    if (MODE == 0) {
      p[o] = c[r];
    } else if (MODE == 1) {
      p[o] = fmaxf(v, 0.0f);
    } else if (MODE == 2) {
      p[o] = 1.0f / (1.0f + expf(-v));
    } else {  // MODE == 3: fused highway gate
      float g = 1.0f / (1.0f + expf(-v));
      p[o] = g * hp[o] + (1.0f - g) * xp[o];
    }
  }
}

// ---------------- kernels ----------------

// dst[n*K + k] = (bf16) src[k*N + n]   (weight transpose+convert, done once)
__global__ void cvt_t_bf16_kernel(const float* __restrict__ src, __bf16* __restrict__ dst,
                                  int K, int N) {
  size_t i = (size_t)blockIdx.x * blockDim.x + threadIdx.x;
  if (i >= (size_t)K * N) return;
  int k = (int)(i / N), n = (int)(i % N);
  dst[(size_t)n * K + k] = (__bf16)src[i];
}

// C[M,N] = Xf32[M,K] @ W, W stored transposed bf16 [N][K].
// 256 threads = 8 waves (4 row-tiles x 2 col-halves); block tile 64x128.
// Explicit 2-stage software pipeline: next K-step fragments are loaded (fresh SSA
// values) BEFORE the current step's WMMAs, so loads overlap XDL within one wave.
// #pragma unroll 2 ping-pongs the buffers, eliminating rotation copies.
template <int MODE>
__global__ __launch_bounds__(256)
void gemm_xf32_bt_kernel(const float* __restrict__ X, const __bf16* __restrict__ Bt,
                         const float* __restrict__ bias, float* __restrict__ C,
                         const float* __restrict__ Hs, const float* __restrict__ Xs,
                         int M, int N, int K) {
  const int w  = threadIdx.x >> 5;
  const int wr = w & 3, wc = w >> 2;
  const int m0 = blockIdx.x * 64 + wr * 16;
  const int n0 = blockIdx.y * 128 + wc * 64;
  v8f acc[4] = {};
  const float*  xrow = X + (size_t)m0 * K;
  const __bf16* bt0  = Bt + (size_t)(n0 +  0) * K;
  const __bf16* bt1  = Bt + (size_t)(n0 + 16) * K;
  const __bf16* bt2  = Bt + (size_t)(n0 + 32) * K;
  const __bf16* bt3  = Bt + (size_t)(n0 + 48) * K;

  // prologue: stage K-step 0
  v16bf a  = frag_a_f32(xrow, K);
  v16bf b0 = frag_b_bf(bt0, K);
  v16bf b1 = frag_b_bf(bt1, K);
  v16bf b2 = frag_b_bf(bt2, K);
  v16bf b3 = frag_b_bf(bt3, K);

#pragma unroll 2
  for (int k0 = 32; k0 < K; k0 += 32) {
    // speculative prefetch of the A stream (global_prefetch_b8; OOB is dropped)
    __builtin_prefetch(xrow + k0 + 224, 0, 0);
    // stage K-step k0 while computing K-step k0-32
    v16bf an  = frag_a_f32(xrow + k0, K);
    v16bf bn0 = frag_b_bf(bt0 + k0, K);
    v16bf bn1 = frag_b_bf(bt1 + k0, K);
    v16bf bn2 = frag_b_bf(bt2 + k0, K);
    v16bf bn3 = frag_b_bf(bt3 + k0, K);
    acc[0] = WMMA_BF16(a, b0, acc[0]);
    acc[1] = WMMA_BF16(a, b1, acc[1]);
    acc[2] = WMMA_BF16(a, b2, acc[2]);
    acc[3] = WMMA_BF16(a, b3, acc[3]);
    a = an; b0 = bn0; b1 = bn1; b2 = bn2; b3 = bn3;
  }
  // drain
  acc[0] = WMMA_BF16(a, b0, acc[0]);
  acc[1] = WMMA_BF16(a, b1, acc[1]);
  acc[2] = WMMA_BF16(a, b2, acc[2]);
  acc[3] = WMMA_BF16(a, b3, acc[3]);

#pragma unroll
  for (int ct = 0; ct < 4; ++ct) {
    size_t off = (size_t)m0 * N + n0 + ct * 16;
    store_c<MODE>(C + off, N,
                  MODE != 0 ? bias + n0 + ct * 16 : (const float*)nullptr,
                  MODE == 3 ? Hs + off : (const float*)nullptr,
                  MODE == 3 ? Xs + off : (const float*)nullptr,
                  acc[ct]);
  }
}

// Batched bf16 GEMM, both operands K-contiguous:
//   A element [m,k] at Am + b*aB + m*aMS + k ; B element [k,n] at Bm + b*bB + n*bNS + k
// Each block: one batch, 8 waves, rows m0=w*16, cols nbase..nbase+127 (8 tiles).
// Same explicit 2-stage software pipeline as above.
__global__ __launch_bounds__(256)
void bgemm_bf16_kernel(const __bf16* __restrict__ Am, long aB, int aMS,
                       const __bf16* __restrict__ Bm, long bB, int bNS,
                       float* __restrict__ C, long cB, int cMS, int K) {
  const int b  = blockIdx.x;
  const int w  = threadIdx.x >> 5;
  const int m0 = w * 16;
  const int nbase = blockIdx.y * 128;
  const __bf16* Ab = Am + (size_t)b * aB + (size_t)m0 * aMS;
  const __bf16* Bb = Bm + (size_t)b * bB + (size_t)nbase * bNS;
  v8f acc[8] = {};

  v16bf a  = frag_a_bf(Ab, aMS);
  v16bf b0 = frag_b_bf(Bb + (size_t)(0 * 16) * bNS, bNS);
  v16bf b1 = frag_b_bf(Bb + (size_t)(1 * 16) * bNS, bNS);
  v16bf b2 = frag_b_bf(Bb + (size_t)(2 * 16) * bNS, bNS);
  v16bf b3 = frag_b_bf(Bb + (size_t)(3 * 16) * bNS, bNS);
  v16bf b4 = frag_b_bf(Bb + (size_t)(4 * 16) * bNS, bNS);
  v16bf b5 = frag_b_bf(Bb + (size_t)(5 * 16) * bNS, bNS);
  v16bf b6 = frag_b_bf(Bb + (size_t)(6 * 16) * bNS, bNS);
  v16bf b7 = frag_b_bf(Bb + (size_t)(7 * 16) * bNS, bNS);

#pragma unroll 2
  for (int k0 = 32; k0 < K; k0 += 32) {
    v16bf an  = frag_a_bf(Ab + k0, aMS);
    v16bf bn0 = frag_b_bf(Bb + (size_t)(0 * 16) * bNS + k0, bNS);
    v16bf bn1 = frag_b_bf(Bb + (size_t)(1 * 16) * bNS + k0, bNS);
    v16bf bn2 = frag_b_bf(Bb + (size_t)(2 * 16) * bNS + k0, bNS);
    v16bf bn3 = frag_b_bf(Bb + (size_t)(3 * 16) * bNS + k0, bNS);
    v16bf bn4 = frag_b_bf(Bb + (size_t)(4 * 16) * bNS + k0, bNS);
    v16bf bn5 = frag_b_bf(Bb + (size_t)(5 * 16) * bNS + k0, bNS);
    v16bf bn6 = frag_b_bf(Bb + (size_t)(6 * 16) * bNS + k0, bNS);
    v16bf bn7 = frag_b_bf(Bb + (size_t)(7 * 16) * bNS + k0, bNS);
    acc[0] = WMMA_BF16(a, b0, acc[0]);
    acc[1] = WMMA_BF16(a, b1, acc[1]);
    acc[2] = WMMA_BF16(a, b2, acc[2]);
    acc[3] = WMMA_BF16(a, b3, acc[3]);
    acc[4] = WMMA_BF16(a, b4, acc[4]);
    acc[5] = WMMA_BF16(a, b5, acc[5]);
    acc[6] = WMMA_BF16(a, b6, acc[6]);
    acc[7] = WMMA_BF16(a, b7, acc[7]);
    a = an;
    b0 = bn0; b1 = bn1; b2 = bn2; b3 = bn3;
    b4 = bn4; b5 = bn5; b6 = bn6; b7 = bn7;
  }
  acc[0] = WMMA_BF16(a, b0, acc[0]);
  acc[1] = WMMA_BF16(a, b1, acc[1]);
  acc[2] = WMMA_BF16(a, b2, acc[2]);
  acc[3] = WMMA_BF16(a, b3, acc[3]);
  acc[4] = WMMA_BF16(a, b4, acc[4]);
  acc[5] = WMMA_BF16(a, b5, acc[5]);
  acc[6] = WMMA_BF16(a, b6, acc[6]);
  acc[7] = WMMA_BF16(a, b7, acc[7]);

  float* Cb = C + (size_t)b * cB + (size_t)m0 * cMS + nbase;
#pragma unroll
  for (int ct = 0; ct < 8; ++ct)
    store_c<0>(Cb + ct * 16, cMS, nullptr, nullptr, nullptr, acc[ct]);
}

// gather rows of layer1: f32 copy, bf16 copy, and bf16 TRANSPOSED copy [B][H][L]
// (the transposed copy is what lets EQ/EA use K-contiguous B fragments)
__global__ void gather_kernel(const float* __restrict__ layer1, const int* __restrict__ idx,
                              float* __restrict__ Of, __bf16* __restrict__ Ob,
                              __bf16* __restrict__ ObT) {
  int r = blockIdx.x, t = threadIdx.x;       // r = b*L + l, t = h
  int b = r >> 7, l = r & (L_ - 1);
  int row = idx[r];
  float v = layer1[(size_t)row * H_ + t];
  Of[(size_t)r * H_ + t] = v;
  Ob[(size_t)r * H_ + t] = (__bf16)v;
  ObT[((size_t)b * H_ + t) * L_ + l] = (__bf16)v;
}

__device__ __forceinline__ float blk128_max(float v, float* sm) {
  int t = threadIdx.x;
#pragma unroll
  for (int off = 16; off >= 1; off >>= 1) v = fmaxf(v, __shfl_xor(v, off, 32));
  if ((t & 31) == 0) sm[t >> 5] = v;
  __syncthreads();
  v = fmaxf(fmaxf(sm[0], sm[1]), fmaxf(sm[2], sm[3]));
  __syncthreads();
  return v;
}
__device__ __forceinline__ float blk128_sum(float v, float* sm) {
  int t = threadIdx.x;
#pragma unroll
  for (int off = 16; off >= 1; off >>= 1) v += __shfl_xor(v, off, 32);
  if ((t & 31) == 0) sm[t >> 5] = v;
  __syncthreads();
  v = sm[0] + sm[1] + sm[2] + sm[3];
  __syncthreads();
  return v;
}

// softmax over last axis (rows); out bf16 [b][q][a]
__global__ void softmax_rows_kernel(const float* __restrict__ S, __bf16* __restrict__ out) {
  __shared__ float sm[4];
  int bq = blockIdx.x, t = threadIdx.x;
  float v = S[(size_t)bq * L_ + t];
  float m = blk128_max(v, sm);
  float e = expf(v - m);
  float s = blk128_sum(e, sm);
  out[(size_t)bq * L_ + t] = (__bf16)(e / s);
}

// softmax over q axis (columns); out stored TRANSPOSED: wab[b][a][q] (bf16)
__global__ void softmax_cols_kernel(const float* __restrict__ S, __bf16* __restrict__ out) {
  __shared__ float sm[4];
  int b = blockIdx.x / L_, a = blockIdx.x % L_, t = threadIdx.x;  // t = q
  float v = S[((size_t)(b * L_ + t)) * L_ + a];
  float m = blk128_max(v, sm);
  float e = expf(v - m);
  float s = blk128_sum(e, sm);
  out[((size_t)(b * L_ + a)) * L_ + t] = (__bf16)(e / s);
}

__global__ void sqdiff_kernel(const float* __restrict__ X, const float* __restrict__ E,
                              float* __restrict__ O, size_t n) {
  size_t i = (size_t)blockIdx.x * blockDim.x + threadIdx.x;
  if (i >= n) return;
  float d = X[i] - E[i];
  O[i] = d * d;
}

// out[b,h] = inv * sum_l X[b,l,h] * (Y ? Y[b,l,h] : 1) ; grid=B, block=H
__global__ void colreduce_kernel(const float* __restrict__ X, const float* __restrict__ Y,
                                 float* __restrict__ out, float inv) {
  int b = blockIdx.x, h = threadIdx.x;
  float acc = 0.0f;
  if (Y) {
    for (int l = 0; l < L_; ++l) {
      size_t i = ((size_t)(b * L_ + l)) * H_ + h;
      acc += X[i] * Y[i];
    }
  } else {
    for (int l = 0; l < L_; ++l)
      acc += X[((size_t)(b * L_ + l)) * H_ + h];
  }
  out[(size_t)b * H_ + h] = acc * inv;
}

// logits = concat(qr,mulq,ar,mula) @ Wlin + blin ; log_softmax over 2
__global__ void final_kernel(const float* __restrict__ qr, const float* __restrict__ mulq,
                             const float* __restrict__ ar, const float* __restrict__ mula,
                             const float* __restrict__ Wlin, const float* __restrict__ blin,
                             float* __restrict__ out) {
  int b = threadIdx.x;
  if (b >= B_) return;
  float a0 = blin[0], a1 = blin[1];
  const float* segs[4] = {qr, mulq, ar, mula};
  for (int s = 0; s < 4; ++s) {
    const float* src = segs[s] + (size_t)b * H_;
    for (int h = 0; h < H_; ++h) {
      float v = src[h];
      int k = s * H_ + h;
      a0 += v * Wlin[2 * k + 0];
      a1 += v * Wlin[2 * k + 1];
    }
  }
  float m = fmaxf(a0, a1);
  float lse = m + logf(expf(a0 - m) + expf(a1 - m));
  out[2 * b + 0] = a0 - lse;
  out[2 * b + 1] = a1 - lse;
}

// ---------------- host launcher ----------------
extern "C" void kernel_launch(void* const* d_in, const int* in_sizes, int n_in,
                              void* d_out, int out_size, void* d_ws, size_t ws_size,
                              hipStream_t stream) {
  const float* A    = (const float*)d_in[0];
  const float* Emb  = (const float*)d_in[1];
  const float* Wh1  = (const float*)d_in[2];
  const float* bh1  = (const float*)d_in[3];
  const float* Wg1  = (const float*)d_in[4];
  const float* bg1  = (const float*)d_in[5];
  const float* Wh2  = (const float*)d_in[6];
  const float* bh2  = (const float*)d_in[7];
  const float* Wg2  = (const float*)d_in[8];
  const float* bg2  = (const float*)d_in[9];
  const float* Wlin = (const float*)d_in[10];
  const float* blin = (const float*)d_in[11];
  const int*   idxq = (const int*)d_in[12];
  const int*   idxa = (const int*)d_in[13];
  float* out = (float*)d_out;

  char* ws = (char*)d_ws;
  size_t off = 0;
  auto alloc = [&](size_t bytes) -> void* {
    void* p = ws + off;
    off = (off + bytes + 255) & ~(size_t)255;
    return p;
  };

  __bf16* Et    = (__bf16*)alloc((size_t)H_ * V_ * 2);
  __bf16* Wh1t  = (__bf16*)alloc((size_t)H_ * H_ * 2);
  __bf16* Wg1t  = (__bf16*)alloc((size_t)H_ * H_ * 2);
  __bf16* Wh2t  = (__bf16*)alloc((size_t)H_ * H_ * 2);
  __bf16* Wg2t  = (__bf16*)alloc((size_t)H_ * H_ * 2);
  float*  layer1= (float*)alloc((size_t)V_ * H_ * 4);
  float*  Qf    = (float*)alloc((size_t)BL_ * H_ * 4);
  float*  Af    = (float*)alloc((size_t)BL_ * H_ * 4);
  __bf16* Qb    = (__bf16*)alloc((size_t)BL_ * H_ * 2);
  __bf16* Abf   = (__bf16*)alloc((size_t)BL_ * H_ * 2);
  __bf16* QbT   = (__bf16*)alloc((size_t)B_ * H_ * L_ * 2);
  __bf16* AbT   = (__bf16*)alloc((size_t)B_ * H_ * L_ * 2);
  float*  Sc    = (float*)alloc((size_t)B_ * L_ * L_ * 4);
  __bf16* wqb   = (__bf16*)alloc((size_t)B_ * L_ * L_ * 2);
  __bf16* wab   = (__bf16*)alloc((size_t)B_ * L_ * L_ * 2);
  float*  EQ    = (float*)alloc((size_t)BL_ * H_ * 4);
  float*  EA    = (float*)alloc((size_t)BL_ * H_ * 4);
  float*  Tb    = (float*)alloc((size_t)BL_ * H_ * 4);
  float*  Hb    = (float*)alloc((size_t)BL_ * H_ * 4);
  float*  X1    = (float*)alloc((size_t)BL_ * H_ * 4);
  float*  H2    = (float*)alloc((size_t)BL_ * H_ * 4);
  float*  qr    = (float*)alloc((size_t)B_ * H_ * 4);
  float*  mulq  = (float*)alloc((size_t)B_ * H_ * 4);
  float*  ar    = (float*)alloc((size_t)B_ * H_ * 4);
  float*  mula  = (float*)alloc((size_t)B_ * H_ * 4);

  const size_t nBLH = (size_t)BL_ * H_;
  const float inv = 1.0f / sqrtf((float)L_);
  const dim3 gW(BL_ / 64, H_ / 128);  // highway GEMM grid

  // --- weight conversions (transposed bf16, [N][K]) ---
  cvt_t_bf16_kernel<<<(V_ * H_ + 255) / 256, 256, 0, stream>>>(Emb, Et, V_, H_);
  cvt_t_bf16_kernel<<<(H_ * H_ + 255) / 256, 256, 0, stream>>>(Wh1, Wh1t, H_, H_);
  cvt_t_bf16_kernel<<<(H_ * H_ + 255) / 256, 256, 0, stream>>>(Wg1, Wg1t, H_, H_);
  cvt_t_bf16_kernel<<<(H_ * H_ + 255) / 256, 256, 0, stream>>>(Wh2, Wh2t, H_, H_);
  cvt_t_bf16_kernel<<<(H_ * H_ + 255) / 256, 256, 0, stream>>>(Wg2, Wg2t, H_, H_);

  // --- GEMM1: layer1 = A @ embedding (A streamed fp32->bf16 in-register) ---
  gemm_xf32_bt_kernel<0><<<dim3(V_ / 64, H_ / 128), 256, 0, stream>>>(
      A, Et, nullptr, layer1, nullptr, nullptr, V_, H_, V_);

  // --- gathers (also emit transposed bf16 copies for EQ/EA B-operands) ---
  gather_kernel<<<BL_, 256, 0, stream>>>(layer1, idxq, Qf, Qb, QbT);
  gather_kernel<<<BL_, 256, 0, stream>>>(layer1, idxa, Af, Abf, AbT);

  // --- score[b] = Q[b] @ Aenc[b]^T  (Aenc row-major [a][h] already acts as Bt [n][k]) ---
  bgemm_bf16_kernel<<<dim3(B_, 1), 256, 0, stream>>>(
      Qb, (long)L_ * H_, H_,
      Abf, (long)L_ * H_, H_,
      Sc, (long)L_ * L_, L_, H_);

  // --- softmaxes ---
  softmax_rows_kernel<<<B_ * L_, 128, 0, stream>>>(Sc, wqb);
  softmax_cols_kernel<<<B_ * L_, 128, 0, stream>>>(Sc, wab);

  // --- EQ = wq @ Aenc ; EA = wa^T @ Q  (B-operands via transposed copies, K-contiguous) ---
  bgemm_bf16_kernel<<<dim3(B_, 2), 256, 0, stream>>>(
      wqb, (long)L_ * L_, L_,
      AbT, (long)H_ * L_, L_,
      EQ, (long)L_ * H_, H_, L_);
  bgemm_bf16_kernel<<<dim3(B_, 2), 256, 0, stream>>>(
      wab, (long)L_ * L_, L_,
      QbT, (long)H_ * L_, L_,
      EA, (long)L_ * H_, H_, L_);

  // --- q path (highway combine fused into the gate GEMM epilogue) ---
  sqdiff_kernel<<<(unsigned)((nBLH + 255) / 256), 256, 0, stream>>>(Qf, EQ, Tb, nBLH);
  colreduce_kernel<<<B_, H_, 0, stream>>>(Qf, EQ, mulq, inv);
  gemm_xf32_bt_kernel<1><<<gW, 256, 0, stream>>>(Tb, Wh1t, bh1, Hb, nullptr, nullptr, BL_, H_, H_);
  gemm_xf32_bt_kernel<3><<<gW, 256, 0, stream>>>(Tb, Wg1t, bg1, X1, Hb, Tb, BL_, H_, H_);
  gemm_xf32_bt_kernel<1><<<gW, 256, 0, stream>>>(X1, Wh2t, bh2, Hb, nullptr, nullptr, BL_, H_, H_);
  gemm_xf32_bt_kernel<3><<<gW, 256, 0, stream>>>(X1, Wg2t, bg2, H2, Hb, X1, BL_, H_, H_);
  colreduce_kernel<<<B_, H_, 0, stream>>>(H2, nullptr, qr, inv);

  // --- a path (reuses Tb/Hb/X1/H2) ---
  sqdiff_kernel<<<(unsigned)((nBLH + 255) / 256), 256, 0, stream>>>(Af, EA, Tb, nBLH);
  colreduce_kernel<<<B_, H_, 0, stream>>>(Af, EA, mula, inv);
  gemm_xf32_bt_kernel<1><<<gW, 256, 0, stream>>>(Tb, Wh1t, bh1, Hb, nullptr, nullptr, BL_, H_, H_);
  gemm_xf32_bt_kernel<3><<<gW, 256, 0, stream>>>(Tb, Wg1t, bg1, X1, Hb, Tb, BL_, H_, H_);
  gemm_xf32_bt_kernel<1><<<gW, 256, 0, stream>>>(X1, Wh2t, bh2, Hb, nullptr, nullptr, BL_, H_, H_);
  gemm_xf32_bt_kernel<3><<<gW, 256, 0, stream>>>(X1, Wg2t, bg2, H2, Hb, X1, BL_, H_, H_);
  colreduce_kernel<<<B_, H_, 0, stream>>>(H2, nullptr, ar, inv);

  // --- final linear + log_softmax ---
  final_kernel<<<1, 64, 0, stream>>>(qr, mulq, ar, mula, Wlin, blin, out);
}
